// SelfAttention_19756849561946
// MI455X (gfx1250) — matrix operand order repaired and checked by reading, not compiled, not run on hardware
//
#include <hip/hip_runtime.h>

// ---------------------------------------------------------------------------
// CDNA5 (gfx1250) self-attention: bf16 WMMA (v_wmma_f32_16x16x32_bf16) +
// Tensor Data Mover (tensor_load_to_lds / s_wait_tensorcnt) pipeline.
// Two-pass flash-style attention: never materializes the 4x4096x4096 scores.
// ---------------------------------------------------------------------------

typedef __attribute__((ext_vector_type(16))) __bf16 v16bf;
typedef __attribute__((ext_vector_type(8)))  __bf16 v8bf;
typedef __attribute__((ext_vector_type(8)))  float  v8f;
typedef __attribute__((ext_vector_type(4)))  unsigned int u32x4;
typedef __attribute__((ext_vector_type(8)))  int i32x8;
typedef __attribute__((ext_vector_type(4)))  int i32x4;

#define WMMA_BF16(a, b, c) \
  __builtin_amdgcn_wmma_f32_16x16x32_bf16(false, (a), false, (b), (short)0, (c), false, false)

__device__ __forceinline__ unsigned short f2bf(float f) {
  unsigned int u = __float_as_uint(f);
  u += 0x7FFFu + ((u >> 16) & 1u);   // round-to-nearest-even
  return (unsigned short)(u >> 16);
}

union V16U { v16bf v; v8bf h[2]; };

// A-fragment (16x32 bf16, MxK): lane<16 holds K {0..15} of row lane%16,
// lane>=16 holds K {16..31}.
__device__ __forceinline__ v16bf ld_a_frag(const unsigned short* lo) {
  V16U u;
  u.h[0] = *(const v8bf*)lo;
  u.h[1] = *(const v8bf*)(lo + 8);
  return u.v;
}

// B-fragment (32x16 bf16, KxN): lane holds 16 contiguous K values of column
// lane%16 (K base 0 for lanes 0-15, 16 for lanes 16-31).
__device__ __forceinline__ v16bf ld_b_frag(const unsigned short* p) {
  V16U u;
  u.h[0] = *(const v8bf*)p;
  u.h[1] = *(const v8bf*)(p + 8);
  return u.v;
}

// ---------------------------------------------------------------------------
// Tensor Data Mover: 2-D bf16 tile load (global -> LDS), D# per ISA ch.8.
//  group0: count=1 | lds_addr | global_addr[56:0] | type=2
//  group1: data_size=2B, tensor_dim0/1 = tile dims (exact tiles, no OOB),
//          tile_dim0 (fastest, contiguous), tile_dim1 (rows), dim0 stride.
// Issue from ONE wave only; completion via s_wait_tensorcnt + barrier.
// ---------------------------------------------------------------------------
__device__ __forceinline__ void tdm_load_2d_bf16(unsigned int lds_off,
                                                 const unsigned short* gptr,
                                                 unsigned int tile_d0,
                                                 unsigned int tile_d1,
                                                 unsigned long long stride0) {
  unsigned long long ga = (unsigned long long)(uintptr_t)gptr;
  u32x4 g0 = { 1u,                                   // count=1 (user D#)
               lds_off,                               // LDS byte address
               (unsigned int)ga,                      // global_addr[31:0]
               ((unsigned int)(ga >> 32) & 0x01FFFFFFu) | 0x80000000u }; // [56:32] | type=2
  i32x8 g1;
  g1[0] = 0x00010000;                                           // data_size=1 (2 bytes)
  g1[1] = (int)((tile_d0 & 0xFFFFu) << 16);                     // tensor_dim0[15:0]
  g1[2] = (int)(((tile_d0 >> 16) & 0xFFFFu) |                   // tensor_dim0[31:16]
                ((tile_d1 & 0xFFFFu) << 16));                   // tensor_dim1[15:0]
  g1[3] = (int)(((tile_d1 >> 16) & 0xFFFFu) |                   // tensor_dim1[31:16]
                ((tile_d0 & 0xFFFFu) << 16));                   // tile_dim0
  g1[4] = (int)(tile_d1 & 0xFFFFu);                             // tile_dim1 (tile_dim2=0)
  g1[5] = (int)(unsigned int)(stride0 & 0xFFFFFFFFu);           // dim0_stride[31:0]
  g1[6] = (int)(unsigned int)((stride0 >> 32) & 0xFFFFu);       // dim0_stride[47:32]
  g1[7] = 0;
  i32x4 gz = {0, 0, 0, 0};
#if __clang_major__ >= 23
  i32x8 gz8 = {0, 0, 0, 0, 0, 0, 0, 0};
  __builtin_amdgcn_tensor_load_to_lds(g0, g1, gz, gz, gz8, 0);
#else
  __builtin_amdgcn_tensor_load_to_lds(g0, g1, gz, gz, 0);
#endif
}

__device__ __forceinline__ unsigned int lds_addr_of(const void* p) {
  return (unsigned int)(uintptr_t)p;
}

// ---------------------------------------------------------------------------
// Kernel 1: f32 -> bf16 convert (x), 4 elements / thread
// ---------------------------------------------------------------------------
__global__ __launch_bounds__(256) void cvt_x_bf16(const float* __restrict__ x,
                                                  unsigned short* __restrict__ xb,
                                                  int n4) {
  int i = blockIdx.x * blockDim.x + threadIdx.x;
  if (i < n4) {
    float4 f = ((const float4*)x)[i];
    unsigned int lo = (unsigned int)f2bf(f.x) | ((unsigned int)f2bf(f.y) << 16);
    unsigned int hi = (unsigned int)f2bf(f.z) | ((unsigned int)f2bf(f.w) << 16);
    ((uint2*)xb)[i] = make_uint2(lo, hi);
  }
}

// ---------------------------------------------------------------------------
// Kernel 2: transpose + convert weights: Wt[n*K + k] = bf16(W[k*N + n])
// grid (N/32, K/32), block (32, 8)
// ---------------------------------------------------------------------------
__global__ __launch_bounds__(256) void transpose_cvt(const float* __restrict__ W,
                                                     unsigned short* __restrict__ Wt,
                                                     int K, int N) {
  __shared__ float tile[32][33];
  int bx = blockIdx.x * 32;  // n
  int by = blockIdx.y * 32;  // k
  int tx = threadIdx.x, ty = threadIdx.y;
#pragma unroll
  for (int i = 0; i < 32; i += 8)
    tile[ty + i][tx] = W[(size_t)(by + ty + i) * N + bx + tx];
  __syncthreads();
#pragma unroll
  for (int i = 0; i < 32; i += 8)
    Wt[(size_t)(bx + ty + i) * K + by + tx] = f2bf(tile[tx][ty + i]);
}

// ---------------------------------------------------------------------------
// Kernel 3: bf16 GEMM  C[M,N] = A[M,K] * Wt[N,K]^T + bias, output bf16.
// TDM double-buffered operand staging; K-step 64 (2 WMMA K-chunks / stage).
// 256 threads = 8 waves; tile 128(M) x 128(N); wave grid 4x2, 2x4 frags/wave.
// transposed==0 : out[m*N+n]; transposed==1 : out[b*N*S + n*S + (m%S)]
// ---------------------------------------------------------------------------
__global__ __launch_bounds__(256) void qkv_gemm(const unsigned short* __restrict__ A,
                                                const unsigned short* __restrict__ Wt,
                                                const float* __restrict__ bias,
                                                unsigned short* __restrict__ out,
                                                int M, int N, int K, int S,
                                                int transposed) {
  __shared__ unsigned short lA[2][128 * 64];
  __shared__ unsigned short lB[2][128 * 64];

  int tid = threadIdx.x;
  int m0 = blockIdx.y * 128;
  int n0 = blockIdx.x * 128;
  int w = tid >> 5;
  int lane = tid & 31;
  int ln16 = lane & 15;
  int hi = lane >> 4;
  int wm = (w >> 1) * 32;   // 0,32,64,96
  int wn = (w & 1) * 64;    // 0,64

  v8f zero = {0.f, 0.f, 0.f, 0.f, 0.f, 0.f, 0.f, 0.f};
  v8f acc[2][4];
#pragma unroll
  for (int mt = 0; mt < 2; ++mt)
#pragma unroll
    for (int nt = 0; nt < 4; ++nt) acc[mt][nt] = zero;

  // prologue: TDM-stage buffer 0 (wave 0 issues; others wait at barrier)
  if (w == 0) {
    tdm_load_2d_bf16(lds_addr_of(&lA[0][0]), A + (size_t)m0 * K, 64, 128,
                     (unsigned long long)K);
    tdm_load_2d_bf16(lds_addr_of(&lB[0][0]), Wt + (size_t)n0 * K, 64, 128,
                     (unsigned long long)K);
    __builtin_amdgcn_s_wait_tensorcnt(0);
  }
  __syncthreads();

  for (int k0 = 0; k0 < K; k0 += 64) {
    int cur = (k0 >> 6) & 1;
    int nxt = cur ^ 1;
    // async prefetch of next stage overlaps with this stage's WMMAs
    if (w == 0 && (k0 + 64) < K) {
      tdm_load_2d_bf16(lds_addr_of(&lA[nxt][0]),
                       A + (size_t)m0 * K + k0 + 64, 64, 128,
                       (unsigned long long)K);
      tdm_load_2d_bf16(lds_addr_of(&lB[nxt][0]),
                       Wt + (size_t)n0 * K + k0 + 64, 64, 128,
                       (unsigned long long)K);
    }
    const unsigned short* pa = &lA[cur][0];
    const unsigned short* pb = &lB[cur][0];
#pragma unroll
    for (int kc = 0; kc < 2; ++kc) {
      v16bf af[2], bf[4];
#pragma unroll
      for (int mt = 0; mt < 2; ++mt) {
        int m = wm + mt * 16 + ln16;
        af[mt] = ld_a_frag(&pa[m * 64 + kc * 32 + hi * 16]);
      }
#pragma unroll
      for (int nt = 0; nt < 4; ++nt) {
        int n = wn + nt * 16 + ln16;
        bf[nt] = ld_b_frag(&pb[n * 64 + kc * 32 + hi * 16]);
      }
#pragma unroll
      for (int mt = 0; mt < 2; ++mt)
#pragma unroll
        for (int nt = 0; nt < 4; ++nt)
          acc[mt][nt] = WMMA_BF16(af[mt], bf[nt], acc[mt][nt]);
    }
    if (w == 0) __builtin_amdgcn_s_wait_tensorcnt(0);
    __syncthreads();
  }

  // epilogue: + bias, convert to bf16, store (optionally transposed)
#pragma unroll
  for (int mt = 0; mt < 2; ++mt) {
#pragma unroll
    for (int nt = 0; nt < 4; ++nt) {
      int col = n0 + wn + nt * 16 + ln16;
      float bv = bias[col];
#pragma unroll
      for (int e = 0; e < 8; ++e) {
        int row = m0 + wm + mt * 16 + e + hi * 8;
        unsigned short v = f2bf(acc[mt][nt][e] + bv);
        if (!transposed) {
          out[(size_t)row * N + col] = v;
        } else {
          int bb = row / S;
          int sl = row - bb * S;
          out[(size_t)bb * N * S + (size_t)col * S + sl] = v;
        }
      }
    }
  }
}

// ---------------------------------------------------------------------------
// Kernel 4: softmax statistics (row max m, row sum l of exp(s-m)).
// One WG = 16 query rows; 8 waves split the 4096 kv positions (16-wide tiles,
// stride 8).  Q tile (16x1024 bf16 = one contiguous 32KB block) staged by a
// single TDM load; K fragments streamed from global (L2-resident).
// ---------------------------------------------------------------------------
__global__ __launch_bounds__(256) void attn_stats(const unsigned short* __restrict__ Qb,
                                                  const unsigned short* __restrict__ Kb,
                                                  float* __restrict__ stm,
                                                  float* __restrict__ stl) {
  constexpr int Sq = 4096, DKc = 1024;
  __shared__ unsigned short lQ[16 * 1024];
  __shared__ float sM[8][16];
  __shared__ float sL[8][16];

  int tid = threadIdx.x;
  int wg = blockIdx.x;          // B * S/16 = 1024
  int b = wg >> 8;
  int q0 = (wg & 255) << 4;

  if (tid < 32) {  // wave 0: one 1-D TDM tile = whole Q block
    tdm_load_2d_bf16(lds_addr_of(&lQ[0]),
                     Qb + (size_t)(b * Sq + q0) * DKc,
                     16384u, 0u, 16384ull);
    __builtin_amdgcn_s_wait_tensorcnt(0);
  }
  __syncthreads();

  int w = tid >> 5;
  int lane = tid & 31;
  int ln16 = lane & 15;
  int hi = lane >> 4;
  const float scale = 0.03125f;  // 1/sqrt(1024)

  float mrun[8], lrun[8];
#pragma unroll
  for (int e = 0; e < 8; ++e) { mrun[e] = -1e30f; lrun[e] = 0.f; }

  v8f zero = {0.f, 0.f, 0.f, 0.f, 0.f, 0.f, 0.f, 0.f};

  for (int t = w; t < 256; t += 8) {
    int j0 = t * 16;
    v8f sacc = zero;
    const unsigned short* krow = Kb + (size_t)(b * Sq + j0 + ln16) * DKc + hi * 16;
#pragma unroll 4
    for (int kc = 0; kc < 32; ++kc) {
      v16bf a = ld_a_frag(&lQ[ln16 * 1024 + kc * 32 + hi * 16]);
      v16bf bfr = ld_b_frag(krow + kc * 32);
      sacc = WMMA_BF16(a, bfr, sacc);
    }
#pragma unroll
    for (int e = 0; e < 8; ++e) {
      float s = sacc[e] * scale;
      float mx = s;
#pragma unroll
      for (int msk = 8; msk >= 1; msk >>= 1) mx = fmaxf(mx, __shfl_xor(mx, msk, 32));
      float mnew = fmaxf(mrun[e], mx);
      float ex = __expf(s - mnew);
#pragma unroll
      for (int msk = 8; msk >= 1; msk >>= 1) ex += __shfl_xor(ex, msk, 32);
      lrun[e] = lrun[e] * __expf(mrun[e] - mnew) + ex;
      mrun[e] = mnew;
    }
  }

  if (ln16 == 0) {
#pragma unroll
    for (int e = 0; e < 8; ++e) {
      sM[w][e + hi * 8] = mrun[e];
      sL[w][e + hi * 8] = lrun[e];
    }
  }
  __syncthreads();
  if (tid < 16) {
    float m = -1e30f;
#pragma unroll
    for (int i = 0; i < 8; ++i) m = fmaxf(m, sM[i][tid]);
    float l = 0.f;
#pragma unroll
    for (int i = 0; i < 8; ++i) l += sL[i][tid] * __expf(sM[i][tid] - m);
    stm[b * Sq + q0 + tid] = m;
    stl[b * Sq + q0 + tid] = l;
  }
}

// ---------------------------------------------------------------------------
// Kernel 5: attention output.  One WG = 16 query rows x full DV.
// Per 128-wide kv block: wave w computes S columns [w*16, w*16+16), writes
// P = exp(S*scale - m) to LDS (16x128 bf16), then each wave accumulates its
// 128-wide DV slice of O += P * V using V stored transposed ([B][DV][S]).
// ---------------------------------------------------------------------------
__global__ __launch_bounds__(256) void attn_out(const unsigned short* __restrict__ Qb,
                                                const unsigned short* __restrict__ Kb,
                                                const unsigned short* __restrict__ Vt,
                                                const float* __restrict__ stm,
                                                const float* __restrict__ stl,
                                                float* __restrict__ out) {
  constexpr int Sq = 4096, DKc = 1024, DVc = 1024;
  __shared__ unsigned short lQ[16 * 1024];
  __shared__ unsigned short lP[16 * 128];

  int tid = threadIdx.x;
  int wg = blockIdx.x;
  int b = wg >> 8;
  int q0 = (wg & 255) << 4;

  if (tid < 32) {  // wave 0: one 1-D TDM tile = whole Q block
    tdm_load_2d_bf16(lds_addr_of(&lQ[0]),
                     Qb + (size_t)(b * Sq + q0) * DKc,
                     16384u, 0u, 16384ull);
    __builtin_amdgcn_s_wait_tensorcnt(0);
  }

  int w = tid >> 5;
  int lane = tid & 31;
  int ln16 = lane & 15;
  int hi = lane >> 4;
  int dvw = w * 128;
  const float scale = 0.03125f;

  float mrow[8], linv[8];
#pragma unroll
  for (int e = 0; e < 8; ++e) {
    int rr = b * Sq + q0 + e + hi * 8;
    mrow[e] = stm[rr];
    linv[e] = 1.0f / stl[rr];
  }

  v8f zero = {0.f, 0.f, 0.f, 0.f, 0.f, 0.f, 0.f, 0.f};
  v8f oacc[8];
#pragma unroll
  for (int nt = 0; nt < 8; ++nt) oacc[nt] = zero;

  __syncthreads();  // lQ ready

  for (int j0 = 0; j0 < Sq; j0 += 128) {
    // ---- S tile: columns j0 + w*16 .. +16, contraction over DK ----
    v8f sacc = zero;
    int jw = j0 + w * 16;
    const unsigned short* krow = Kb + (size_t)(b * Sq + jw + ln16) * DKc + hi * 16;
#pragma unroll 4
    for (int kc = 0; kc < 32; ++kc) {
      v16bf a = ld_a_frag(&lQ[ln16 * 1024 + kc * 32 + hi * 16]);
      v16bf bfr = ld_b_frag(krow + kc * 32);
      sacc = WMMA_BF16(a, bfr, sacc);
    }

    __syncthreads();  // previous P consumers done
#pragma unroll
    for (int e = 0; e < 8; ++e) {
      float p = __expf(sacc[e] * scale - mrow[e]);
      int row = e + hi * 8;
      lP[row * 128 + w * 16 + ln16] = f2bf(p);
    }
    __syncthreads();  // P ready

    // ---- O += P(16x128) * V(128 x dv-slice), V from transposed layout ----
#pragma unroll
    for (int kc = 0; kc < 4; ++kc) {
      v16bf a = ld_a_frag(&lP[ln16 * 128 + kc * 32 + hi * 16]);
#pragma unroll
      for (int nt = 0; nt < 8; ++nt) {
        int n = dvw + nt * 16 + ln16;
        const unsigned short* vcol =
            Vt + (size_t)(b * DVc + n) * Sq + j0 + kc * 32 + hi * 16;
        v16bf bfr = ld_b_frag(vcol);
        oacc[nt] = WMMA_BF16(a, bfr, oacc[nt]);
      }
    }
  }

  // ---- epilogue: normalize by l, write f32 output ----
#pragma unroll
  for (int nt = 0; nt < 8; ++nt) {
    int col = dvw + nt * 16 + ln16;
#pragma unroll
    for (int e = 0; e < 8; ++e) {
      int row = e + hi * 8;
      out[(size_t)(b * Sq + q0 + row) * DVc + col] = oacc[nt][e] * linv[e];
    }
  }
}

// ---------------------------------------------------------------------------
// Host-side launcher
// ---------------------------------------------------------------------------
extern "C" void kernel_launch(void* const* d_in, const int* in_sizes, int n_in,
                              void* d_out, int out_size, void* d_ws, size_t ws_size,
                              hipStream_t stream) {
  (void)in_sizes; (void)n_in; (void)out_size; (void)ws_size;

  const int B = 4, S = 4096, D = 1024, DK = 1024, DV = 1024;
  const int M = B * S;  // 16384

  const float* x  = (const float*)d_in[0];
  const float* Wq = (const float*)d_in[1];
  const float* bq = (const float*)d_in[2];
  const float* Wk = (const float*)d_in[3];
  const float* bk = (const float*)d_in[4];
  const float* Wv = (const float*)d_in[5];
  const float* bv = (const float*)d_in[6];
  float* out = (float*)d_out;

  char* ws = (char*)d_ws;
  unsigned short* xb  = (unsigned short*)ws; ws += (size_t)M * D * 2;    // 32 MB
  unsigned short* Wqt = (unsigned short*)ws; ws += (size_t)D * DK * 2;   // 2 MB
  unsigned short* Wkt = (unsigned short*)ws; ws += (size_t)D * DK * 2;
  unsigned short* Wvt = (unsigned short*)ws; ws += (size_t)D * DV * 2;
  unsigned short* Qb  = (unsigned short*)ws; ws += (size_t)M * DK * 2;   // 32 MB
  unsigned short* Kb  = (unsigned short*)ws; ws += (size_t)M * DK * 2;   // 32 MB
  unsigned short* Vt  = (unsigned short*)ws; ws += (size_t)M * DV * 2;   // 32 MB (B,DV,S)
  float* stm = (float*)ws; ws += (size_t)B * S * 4;
  float* stl = (float*)ws; ws += (size_t)B * S * 4;

  // 1. x -> bf16
  {
    int n4 = M * D / 4;
    cvt_x_bf16<<<(n4 + 255) / 256, 256, 0, stream>>>(x, xb, n4);
  }
  // 2. weights -> bf16 transposed (N-major for contiguous-K B fragments)
  {
    dim3 blk(32, 8);
    dim3 grd(DK / 32, D / 32);
    transpose_cvt<<<grd, blk, 0, stream>>>(Wq, Wqt, D, DK);
    transpose_cvt<<<grd, blk, 0, stream>>>(Wk, Wkt, D, DK);
    transpose_cvt<<<grd, blk, 0, stream>>>(Wv, Wvt, D, DV);
  }
  // 3. Q, K (row-major bf16), V (transposed bf16)
  {
    dim3 grd(DK / 128, M / 128);
    qkv_gemm<<<grd, 256, 0, stream>>>(xb, Wqt, bq, Qb, M, DK, D, S, 0);
    qkv_gemm<<<grd, 256, 0, stream>>>(xb, Wkt, bk, Kb, M, DK, D, S, 0);
    qkv_gemm<<<grd, 256, 0, stream>>>(xb, Wvt, bv, Vt, M, DV, D, S, 1);
  }
  // 4. softmax stats (m, l) per query row
  attn_stats<<<B * S / 16, 256, 0, stream>>>(Qb, Kb, stm, stl);
  // 5. attention output
  attn_out<<<B * S / 16, 256, 0, stream>>>(Qb, Kb, Vt, stm, stl, out);
}
